// NormLocalAttention_14671608283287
// MI455X (gfx1250) — matrix operand order repaired and checked by reading, compiled, not run on hardware
//
#include <hip/hip_runtime.h>
#include <hip/hip_bf16.h>

// ---------------------------------------------------------------------------
// NormLocalAttention for MI455X (gfx1250, wave32, WMMA + Tensor Data Mover)
// bf16 WMMA (v_wmma_f32_16x16x32_bf16) for all GEMMs, fp32 accumulate.
// Shared tiles (weights, K/V panels) staged into LDS by the TDM
// (tensor_load_to_lds), double-buffered against the WMMA stream.
// ---------------------------------------------------------------------------

typedef __attribute__((ext_vector_type(16))) __bf16 v16bf;
typedef __attribute__((ext_vector_type(8)))  __bf16 v8bf;
typedef __attribute__((ext_vector_type(8)))  float  v8f;
typedef __attribute__((ext_vector_type(4)))  float  v4f;
typedef __attribute__((ext_vector_type(4)))  unsigned int u32x4;
typedef __attribute__((ext_vector_type(8)))  int    i32x8;
typedef __attribute__((ext_vector_type(4)))  int    i32x4;

#define BQ   2
#define SEQ  2048
#define EMB  1024
#define NH   16
#define HD   64
#define MTOT (BQ * SEQ)   // 4096

union V16 { v16bf v; v8bf h[2]; };

__device__ __forceinline__ v8f wmma_bf16(v16bf a, v16bf b, v8f c) {
    // 8 args: (neg_a, A, neg_b, B, c_mod, C, reuse_a, reuse_b)
    return __builtin_amdgcn_wmma_f32_16x16x32_bf16(false, a, false, b,
                                                   (short)0, c, false, false);
}

// ---------------------------------------------------------------------------
// Tensor Data Mover: async 2D bf16 tile load Global -> LDS.
// D# packing per CDNA5 ISA ch.8 (group0: count/lds/global/type,
// group1: data_size=2B, tensor dims, tile dims, dim0 stride).
// This toolchain exposes the 6-arg builtin:
//   (u32x4 g0, i32x8 g1, i32x4 g2, i32x4 g3, i32x8 g4, i32 cpol)
// Issue from ONE wave only; completion via S_WAIT_TENSORCNT.
// ---------------------------------------------------------------------------
__device__ __forceinline__ void tdm_load_2d_bf16(unsigned lds_addr, const void* gptr,
                                                 unsigned tile_w, unsigned tile_h,
                                                 unsigned stride_elems,
                                                 unsigned tensor_w, unsigned tensor_h) {
    unsigned long long ga = (unsigned long long)(size_t)gptr;
    u32x4 g0;
    g0[0] = 1u;                                   // count=1, user mode, no gather
    g0[1] = lds_addr;                             // LDS byte address
    g0[2] = (unsigned)ga;                         // global_addr[31:0]
    g0[3] = (unsigned)((ga >> 32) & 0x01FFFFFFu)  // global_addr[56:32]
          | (2u << 30);                           // type = 2 ("image")
    i32x8 g1;
    g1[0] = (int)(1u << 16);                      // data_size = 1 -> 2 bytes
    g1[1] = (int)((tensor_w & 0xFFFFu) << 16);    // tensor_dim0[15:0]
    g1[2] = (int)(((tensor_w >> 16) & 0xFFFFu) | ((tensor_h & 0xFFFFu) << 16));
    g1[3] = (int)(((tensor_h >> 16) & 0xFFFFu) | ((tile_w & 0xFFFFu) << 16));
    g1[4] = (int)(tile_h & 0xFFFFu);              // tile_dim1 (tile_dim2 = 0)
    g1[5] = (int)stride_elems;                    // tensor_dim0_stride[31:0]
    g1[6] = 0;                                    // stride hi / dim1_stride lo
    g1[7] = 0;
    i32x4 g2 = {1, 0, 0, 0};                      // tensor_dim2 = 1, rest unused
    i32x4 g3 = {0, 0, 0, 0};
    i32x8 g4 = {0, 0, 0, 0, 0, 0, 0, 0};          // unused trailing group
    __builtin_amdgcn_tensor_load_to_lds(g0, g1, g2, g3, g4, 0);
}

__device__ __forceinline__ unsigned lds_off(const void* p) {
    // flat shared address truncates to the LDS byte offset (ISA 10.2);
    // also escapes the pointer so LDS loads after TDM writes stay live.
    return (unsigned)(size_t)p;
}

// ---------------------------------------------------------------------------
// fp32 -> bf16 cast (vectorized x4; all sizes here are multiples of 4)
// ---------------------------------------------------------------------------
__global__ void cast_f32_bf16(const float* __restrict__ src,
                              __bf16* __restrict__ dst, int n4) {
    int i = blockIdx.x * blockDim.x + threadIdx.x;
    if (i < n4) {
        v4f x = *(const v4f*)(src + i * 4);
        __bf16* d = dst + i * 4;
        d[0] = (__bf16)x[0]; d[1] = (__bf16)x[1];
        d[2] = (__bf16)x[2]; d[3] = (__bf16)x[3];
    }
}

// ---------------------------------------------------------------------------
// GEMM: out = act(X[M,K] @ W[N,K]^T + bias[N])
//   wave tile 16x64 (4 C accumulators), block = 8 waves stacked in M (128x64)
//   Weight tile (64 rows x 32 k, shared by all 8 waves) is TDM-staged into
//   LDS, double-buffered.
//   ACT: 0 = none, 1 = SiLU
//   SMODE: 0 = fp32 flat [M][EMB]
//          1 = bf16 head-split  [B*NH][SEQ][HD]
//          2 = bf16 head-split, transposed per head [B*NH][HD][SEQ]
// ---------------------------------------------------------------------------
template <int ACT, int SMODE>
__global__ void gemm_e(const __bf16* __restrict__ X, const __bf16* __restrict__ W,
                       const float* __restrict__ bias, void* __restrict__ dst) {
    __shared__ __align__(16) __bf16 wt[2][64 * 32];   // 2 x 4KB weight tiles

    const int lane  = threadIdx.x & 31;
    const int wave  = threadIdx.x >> 5;
    const int lm    = lane & 15;      // column / row-in-tile index
    const int lh    = lane >> 4;      // lane-half select
    const int mBase = (blockIdx.x * 8 + wave) * 16;
    const int nBase = blockIdx.y * 64;

    v8f acc[4] = {};

    // A-fragment source row (16-bit A 16x32 layout)
    const __bf16* xrow = X + (size_t)(mBase + lm) * EMB + lh * 8;
    const __bf16* wbase = W + (size_t)nBase * EMB;

    // prologue: stage first weight tile
    if (wave == 0)
        tdm_load_2d_bf16(lds_off(&wt[0][0]), wbase, 32, 64, EMB, EMB, EMB);

    for (int kb = 0; kb < EMB; kb += 32) {
        const int cur = (kb >> 5) & 1;
        if (wave == 0) __builtin_amdgcn_s_wait_tensorcnt(0);
        __syncthreads();                      // tile[cur] ready for everyone
        if (wave == 0 && kb + 32 < EMB)
            tdm_load_2d_bf16(lds_off(&wt[cur ^ 1][0]), wbase + kb + 32,
                             32, 64, EMB, EMB, EMB);

        V16 a;
        a.h[0] = *(const v8bf*)(xrow + kb);
        a.h[1] = *(const v8bf*)(xrow + kb + 16);
        const __bf16* wl = &wt[cur][0];
#pragma unroll
        for (int t = 0; t < 4; ++t) {
            // B-fragment (32x16): lane = N (tile row), K = 8h..8h+7, 16+8h..16+8h+7
            const __bf16* wr = wl + (t * 16 + lm) * 32 + lh * 8;
            V16 b;
            b.h[0] = *(const v8bf*)(wr);
            b.h[1] = *(const v8bf*)(wr + 16);
            acc[t] = wmma_bf16(a.v, b.v, acc[t]);
        }
        __syncthreads();                      // reads of tile[cur] done
    }

    // Epilogue. C/D v8f layout: row = r + lh*8, col = lm (per 16x16 tile)
#pragma unroll
    for (int t = 0; t < 4; ++t) {
        const int n = nBase + t * 16 + lm;
        const float bn = bias[n];
#pragma unroll
        for (int r = 0; r < 8; ++r) {
            const int m = mBase + r + lh * 8;
            float v = acc[t][r] + bn;
            if (ACT == 1) v = v / (1.0f + __expf(-v));   // SiLU
            if (SMODE == 0) {
                ((float*)dst)[(size_t)m * EMB + n] = v;
            } else {
                const int b  = m >> 11;          // m / SEQ
                const int ns = m & (SEQ - 1);
                const int h  = n >> 6;           // n / HD
                const int d  = n & (HD - 1);
                if (SMODE == 1)
                    ((__bf16*)dst)[(((size_t)(b * NH + h) * SEQ + ns) * HD) + d] = (__bf16)v;
                else
                    ((__bf16*)dst)[(((size_t)(b * NH + h) * HD + d) * SEQ) + ns] = (__bf16)v;
            }
        }
    }
}

// ---------------------------------------------------------------------------
// Attention: O[bh] = relu(Q[bh] K[bh]^T / 8) @ V[bh]
//   Q,K: [BH][SEQ][HD] bf16; Vt: [BH][HD][SEQ] bf16; O: [BH][SEQ][HD] fp32
//   One wave = 16 query rows; loop key columns in chunks of 32.
//   K panel (32x64, contiguous) and V panel (64x32, stride SEQ) are shared by
//   all 8 waves -> TDM-staged into LDS, double-buffered.
//   S (f32 C-layout) -> bf16 A-layout via per-wave LDS bounce.
// ---------------------------------------------------------------------------
__global__ void attn_kernel(const __bf16* __restrict__ Q, const __bf16* __restrict__ K,
                            const __bf16* __restrict__ Vt, float* __restrict__ O) {
    __shared__ __align__(16) __bf16 sbounce[8][16 * 32];  // 8KB: S shuffle
    __shared__ __align__(16) __bf16 kt[2][32 * 64];       // 8KB: K panels
    __shared__ __align__(16) __bf16 vt[2][64 * 32];       // 8KB: V panels

    const int lane  = threadIdx.x & 31;
    const int wave  = threadIdx.x >> 5;
    const int lm    = lane & 15;
    const int lh    = lane >> 4;
    const int bh    = blockIdx.y;                       // 0..BQ*NH-1
    const int mBase = blockIdx.x * 128 + wave * 16;

    const __bf16* Qb = Q  + (size_t)bh * SEQ * HD;
    const __bf16* Kb = K  + (size_t)bh * SEQ * HD;
    const __bf16* Vb = Vt + (size_t)bh * HD * SEQ;

    // Q A-fragments for 16 rows x HD=64 (two 16x32 fragments), kept in VGPRs
    const __bf16* qrow = Qb + (size_t)(mBase + lm) * HD + lh * 8;
    V16 aq0, aq1;
    aq0.h[0] = *(const v8bf*)(qrow +  0);
    aq0.h[1] = *(const v8bf*)(qrow + 16);
    aq1.h[0] = *(const v8bf*)(qrow + 32);
    aq1.h[1] = *(const v8bf*)(qrow + 48);

    v8f o[4] = {};
    __bf16* sl = &sbounce[wave][0];

    // prologue: stage panels for j = 0
    if (wave == 0) {
        tdm_load_2d_bf16(lds_off(&kt[0][0]), Kb, 64, 32, HD, HD, SEQ);
        tdm_load_2d_bf16(lds_off(&vt[0][0]), Vb, 32, 64, SEQ, SEQ, HD);
    }

    for (int j = 0; j < SEQ; j += 32) {
        const int cur = (j >> 5) & 1;
        if (wave == 0) __builtin_amdgcn_s_wait_tensorcnt(0);
        __syncthreads();                              // panels[cur] ready
        if (wave == 0 && j + 32 < SEQ) {
            tdm_load_2d_bf16(lds_off(&kt[cur ^ 1][0]), Kb + (size_t)(j + 32) * HD,
                             64, 32, HD, HD, SEQ);
            tdm_load_2d_bf16(lds_off(&vt[cur ^ 1][0]), Vb + (j + 32),
                             32, 64, SEQ, SEQ, HD);
        }

        // ---- S = Q K^T over this 32-column chunk (two 16x16 tiles) ----
        const __bf16* kl = &kt[cur][0];
        v8f s[2] = {};
#pragma unroll
        for (int t = 0; t < 2; ++t) {
            // B[k=d][n] = K[j + n][d]; lane = n (panel row), d = 8h..8h+7, 16+8h..16+8h+7
            const __bf16* kr = kl + (t * 16 + lm) * HD + lh * 8;
            V16 b0, b1;
            b0.h[0] = *(const v8bf*)(kr);       b0.h[1] = *(const v8bf*)(kr + 16);
            b1.h[0] = *(const v8bf*)(kr + 32);  b1.h[1] = *(const v8bf*)(kr + 48);
            s[t] = wmma_bf16(aq0.v, b0.v, s[t]);  // d = 0..31
            s[t] = wmma_bf16(aq1.v, b1.v, s[t]);  // d = 32..63
        }
        // ---- relu, scale 1/sqrt(64), C-layout -> row-major bf16 in LDS ----
#pragma unroll
        for (int t = 0; t < 2; ++t)
#pragma unroll
            for (int r = 0; r < 8; ++r) {
                float v = fmaxf(s[t][r], 0.0f) * 0.125f;
                sl[(r + lh * 8) * 32 + t * 16 + lm] = (__bf16)v;
            }
        asm volatile("s_wait_dscnt 0" ::: "memory");   // same-wave LDS RAW
        // ---- read back as one 16x32 A-fragment ----
        V16 as;
        as.h[0] = *(const v8bf*)(sl + lm * 32 + lh * 8);
        as.h[1] = *(const v8bf*)(sl + lm * 32 + lh * 8 + 16);
        // ---- O += S @ V ; B[k=seq][n=d] = Vt[d][seq] (panel row = d) ----
        const __bf16* vl = &vt[cur][0];
#pragma unroll
        for (int t = 0; t < 4; ++t) {
            const __bf16* vr = vl + (t * 16 + lm) * 32 + lh * 8;
            V16 bv;
            bv.h[0] = *(const v8bf*)(vr);
            bv.h[1] = *(const v8bf*)(vr + 16);
            o[t] = wmma_bf16(as.v, bv.v, o[t]);
        }
        __syncthreads();                              // reads of panels[cur] done
    }

    float* Ob = O + (size_t)bh * SEQ * HD;
#pragma unroll
    for (int t = 0; t < 4; ++t)
#pragma unroll
        for (int r = 0; r < 8; ++r)
            Ob[(size_t)(mBase + r + lh * 8) * HD + t * 16 + lm] = o[t][r];
}

// ---------------------------------------------------------------------------
// Merge heads + LayerNorm(E) + gate by U, emit bf16 Y for the final GEMM.
// One block (256 threads) per row m; 4 features per thread.
// ---------------------------------------------------------------------------
__global__ void ln_gate_kernel(const float* __restrict__ O, const float* __restrict__ U,
                               const float* __restrict__ g, const float* __restrict__ be,
                               __bf16* __restrict__ Y) {
    __shared__ float red[256];
    const int m   = blockIdx.x;
    const int tid = threadIdx.x;
    const int b   = m >> 11;
    const int ns  = m & (SEQ - 1);
    const int e0  = tid * 4;
    const int h   = e0 >> 6;
    const int d   = e0 & (HD - 1);

    const float* src = O + (((size_t)(b * NH + h) * SEQ + ns) * HD) + d;
    v4f x = *(const v4f*)src;

    float s  = x[0] + x[1] + x[2] + x[3];
    float s2 = x[0] * x[0] + x[1] * x[1] + x[2] * x[2] + x[3] * x[3];

    red[tid] = s;  __syncthreads();
    for (int off = 128; off > 0; off >>= 1) {
        if (tid < off) red[tid] += red[tid + off];
        __syncthreads();
    }
    const float mean = red[0] * (1.0f / EMB);
    __syncthreads();
    red[tid] = s2; __syncthreads();
    for (int off = 128; off > 0; off >>= 1) {
        if (tid < off) red[tid] += red[tid + off];
        __syncthreads();
    }
    const float var  = red[0] * (1.0f / EMB) - mean * mean;
    const float rstd = rsqrtf(var + 1e-5f);

#pragma unroll
    for (int i = 0; i < 4; ++i) {
        const int e = e0 + i;
        float v = (x[i] - mean) * rstd * g[e] + be[e];
        v *= U[(size_t)m * EMB + e];
        Y[(size_t)m * EMB + e] = (__bf16)v;
    }
}

// ---------------------------------------------------------------------------
// Launcher
// inputs: 0:x 1:Wq 2:bq 3:Wk 4:bk 5:Wv 6:bv 7:Wu 8:bu 9:Wo 10:bo 11:ln_g 12:ln_b
// ---------------------------------------------------------------------------
extern "C" void kernel_launch(void* const* d_in, const int* in_sizes, int n_in,
                              void* d_out, int out_size, void* d_ws, size_t ws_size,
                              hipStream_t stream) {
    (void)in_sizes; (void)n_in; (void)out_size; (void)ws_size;

    const float* x    = (const float*)d_in[0];
    const float* Wq   = (const float*)d_in[1];
    const float* bq   = (const float*)d_in[2];
    const float* Wk   = (const float*)d_in[3];
    const float* bk   = (const float*)d_in[4];
    const float* Wv   = (const float*)d_in[5];
    const float* bv   = (const float*)d_in[6];
    const float* Wu   = (const float*)d_in[7];
    const float* bu   = (const float*)d_in[8];
    const float* Wo   = (const float*)d_in[9];
    const float* bo   = (const float*)d_in[10];
    const float* ln_g = (const float*)d_in[11];
    const float* ln_b = (const float*)d_in[12];

    char* ws = (char*)d_ws;
    size_t off = 0;
    auto carve = [&](size_t bytes) { char* p = ws + off; off += (bytes + 255) & ~(size_t)255; return p; };

    __bf16* xb  = (__bf16*)carve((size_t)MTOT * EMB * 2);   // x bf16
    __bf16* wqb = (__bf16*)carve((size_t)EMB * EMB * 2);
    __bf16* wkb = (__bf16*)carve((size_t)EMB * EMB * 2);
    __bf16* wvb = (__bf16*)carve((size_t)EMB * EMB * 2);
    __bf16* wub = (__bf16*)carve((size_t)EMB * EMB * 2);
    __bf16* wob = (__bf16*)carve((size_t)EMB * EMB * 2);
    __bf16* Qh  = (__bf16*)carve((size_t)MTOT * EMB * 2);   // [BH][SEQ][HD]
    __bf16* Kh  = (__bf16*)carve((size_t)MTOT * EMB * 2);   // [BH][SEQ][HD]
    __bf16* Vtr = (__bf16*)carve((size_t)MTOT * EMB * 2);   // [BH][HD][SEQ]
    float*  Uw  = (float*) carve((size_t)MTOT * EMB * 4);   // silu(u), fp32
    float*  Ow  = (float*) carve((size_t)MTOT * EMB * 4);   // attention out fp32
    __bf16* Yw  = (__bf16*)carve((size_t)MTOT * EMB * 2);   // gated/normed bf16

    // 1) casts
    {
        const int nx = MTOT * EMB / 4, nw = EMB * EMB / 4, thr = 256;
        cast_f32_bf16<<<(nx + thr - 1) / thr, thr, 0, stream>>>(x,  xb,  nx);
        cast_f32_bf16<<<(nw + thr - 1) / thr, thr, 0, stream>>>(Wq, wqb, nw);
        cast_f32_bf16<<<(nw + thr - 1) / thr, thr, 0, stream>>>(Wk, wkb, nw);
        cast_f32_bf16<<<(nw + thr - 1) / thr, thr, 0, stream>>>(Wv, wvb, nw);
        cast_f32_bf16<<<(nw + thr - 1) / thr, thr, 0, stream>>>(Wu, wub, nw);
        cast_f32_bf16<<<(nw + thr - 1) / thr, thr, 0, stream>>>(Wo, wob, nw);
    }

    // 2) projections: M=4096, N=1024 -> grid (M/128, N/64)
    const dim3 gproj(MTOT / 128, EMB / 64), blk(256);
    gemm_e<0, 1><<<gproj, blk, 0, stream>>>(xb, wqb, bq, Qh);    // Q
    gemm_e<0, 1><<<gproj, blk, 0, stream>>>(xb, wkb, bk, Kh);    // K
    gemm_e<1, 2><<<gproj, blk, 0, stream>>>(xb, wvb, bv, Vtr);   // V = silu, transposed
    gemm_e<1, 0><<<gproj, blk, 0, stream>>>(xb, wub, bu, Uw);    // U = silu, fp32

    // 3) attention: grid (SEQ/128, B*H)
    attn_kernel<<<dim3(SEQ / 128, BQ * NH), blk, 0, stream>>>(Qh, Kh, Vtr, Ow);

    // 4) merge + layernorm + gate
    ln_gate_kernel<<<MTOT, 256, 0, stream>>>(Ow, Uw, ln_g, ln_b, Yw);

    // 5) output projection -> d_out (fp32)
    gemm_e<0, 0><<<gproj, blk, 0, stream>>>(Yw, wob, bo, d_out);
}
